// StandardMultiHeadAttention_5257039970890
// MI455X (gfx1250) — matrix operand order repaired and checked by reading, not compiled, not run on hardware
//
#include <hip/hip_runtime.h>
#include <hip/hip_bf16.h>

// ---------------------------------------------------------------------------
// MHA forward for MI455X (gfx1250): bf16 WMMA GEMMs + fused flash attention.
//   1) cast x, Wq, Wk, Wv, Wo -> bf16 in d_ws
//   2) QKV projection: LDS-staged GEMM (async-to-LDS DMA, double-buffered),
//      Q,K -> [B,H,S,64]; V -> V^T [B,H,64,S]
//   3) flash attention: K/V chunks async-staged to LDS, WMMA scores/PV,
//      online softmax in f32, scores never touch HBM
//   4) output projection (same LDS-staged GEMM) -> f32 d_out
// ---------------------------------------------------------------------------

typedef __bf16 bf16;
typedef __attribute__((ext_vector_type(16))) __bf16 v16bf;
typedef __attribute__((ext_vector_type(8)))  __bf16 v8bf;
typedef __attribute__((ext_vector_type(8)))  float  v8f;

#define DM 1024     // d_model
#define NH 16       // heads
#define DH 64       // head dim
#define BB 4        // batch
#define SS 2048     // seq
#define MT (BB*SS)  // 8192 rows
#define LP 40       // padded LDS row stride (20 dwords -> conflict-free)

static __device__ __forceinline__ v8f vzero8() {
  v8f z;
#pragma unroll
  for (int i = 0; i < 8; ++i) z[i] = 0.0f;
  return z;
}

// WMMA A fragment (16x32 bf16, lane = row m): elements 0..7 -> K = half*8+e,
// elements 8..15 -> K = 16 + half*8 + e  (two contiguous 16B chunks).
static __device__ __forceinline__ v16bf frag_a(const bf16* base, int half) {
  const v8bf lo = *reinterpret_cast<const v8bf*>(base + half * 8);
  const v8bf hi = *reinterpret_cast<const v8bf*>(base + 16 + half * 8);
  v16bf r;
#pragma unroll
  for (int i = 0; i < 8; ++i) { r[i] = lo[i]; r[i + 8] = hi[i]; }
  return r;
}

// WMMA B fragment (32x16 bf16, lane = col n): elements 0..7 -> K = half*8+e,
// elements 8..15 -> K = 16 + half*8 + e (same order as frag_a).
static __device__ __forceinline__ v16bf frag_b(const bf16* base, int half) {
  const v8bf lo = *reinterpret_cast<const v8bf*>(base + half * 8);
  const v8bf hi = *reinterpret_cast<const v8bf*>(base + 16 + half * 8);
  v16bf r;
#pragma unroll
  for (int i = 0; i < 8; ++i) { r[i] = lo[i]; r[i + 8] = hi[i]; }
  return r;
}

static __device__ __forceinline__ v8f wmma_bf16(v16bf a, v16bf b, v8f c) {
  return __builtin_amdgcn_wmma_f32_16x16x32_bf16(false, a, false, b, (short)0, c,
                                                 false, false);
}

// Async 16B global -> LDS copy (per lane). LDS generic pointers carry the
// wave-relative LDS offset in addr[31:0] (ISA 10.2 aperture rules).
static __device__ __forceinline__ void async_ld16(const bf16* lds_dst,
                                                  const bf16* gsrc) {
  const unsigned off = (unsigned)(unsigned long long)lds_dst;
  asm volatile("global_load_async_to_lds_b128 %0, %1, off"
               :
               : "v"(off), "v"(gsrc)
               : "memory");
}

static __device__ __forceinline__ void wait_async0() {
  asm volatile("s_wait_asynccnt 0x0" ::: "memory");
}

// Stage a 128-row x 32-K bf16 tile (row stride LP in LDS, DM in global).
// 512 x 16B chunks, 2 per thread.
static __device__ __forceinline__ void stage_tile(bf16 (*dst)[LP],
                                                  const bf16* src, int tid) {
#pragma unroll
  for (int i = 0; i < 2; ++i) {
    const int c   = tid + i * 256;  // 0..511
    const int row = c >> 2;
    const int cc  = (c & 3) * 8;
    async_ld16(&dst[row][cc], src + (size_t)row * DM + cc);
  }
}

// ---------------------------------------------------------------------------
__global__ void cast_f32_to_bf16(const float* __restrict__ src,
                                 bf16* __restrict__ dst, int n) {
  int i = blockIdx.x * blockDim.x + threadIdx.x;
  int stride = gridDim.x * blockDim.x;
  for (; i < n; i += stride) dst[i] = (bf16)src[i];
}

// ---------------------------------------------------------------------------
// QKV projection. grid = (MT/128, DM/128, 3). block = 256 (8 waves).
// Block tile 128x128, LDS double-buffered via async DMA; wave tile 32x64.
__global__ void __launch_bounds__(256) qkv_gemm_kernel(
    const bf16* __restrict__ X,
    const bf16* __restrict__ Wq, const bf16* __restrict__ Wk,
    const bf16* __restrict__ Wv,
    const float* __restrict__ bq, const float* __restrict__ bk,
    const float* __restrict__ bv,
    bf16* __restrict__ Qo, bf16* __restrict__ Ko, bf16* __restrict__ VTo) {
  __shared__ __align__(16) bf16 At[2][128][LP];
  __shared__ __align__(16) bf16 Bt[2][128][LP];

  const int tid  = threadIdx.x;
  const int lane = tid & 31;
  const int wv   = tid >> 5;
  const int half = lane >> 4;
  const int lm   = lane & 15;
  const int mode = blockIdx.z;  // 0=Q, 1=K, 2=V^T

  const bf16*  W    = (mode == 0) ? Wq : (mode == 1) ? Wk : Wv;
  const float* bias = (mode == 0) ? bq : (mode == 1) ? bk : bv;

  const int mwave = (wv & 3) * 32;   // wave M offset in block tile
  const int nwave = (wv >> 2) * 64;  // wave N offset in block tile
  const int mbase = blockIdx.x * 128 + mwave;
  const int nbase = blockIdx.y * 128 + nwave;

  const bf16* Ablk = X + (size_t)(blockIdx.x * 128) * DM;
  const bf16* Bblk = W + (size_t)(blockIdx.y * 128) * DM;

  stage_tile(At[0], Ablk, tid);
  stage_tile(Bt[0], Bblk, tid);

  v8f acc[2][4];
#pragma unroll
  for (int g = 0; g < 2; ++g)
#pragma unroll
    for (int t = 0; t < 4; ++t) acc[g][t] = vzero8();

  for (int k0 = 0; k0 < DM; k0 += 32) {
    const int cur = (k0 >> 5) & 1;

    wait_async0();      // our share of buffer `cur` has landed
    __syncthreads();    // everyone's share has landed; prior readers done

    if (k0 + 32 < DM) {
      const int nxt = cur ^ 1;
      stage_tile(At[nxt], Ablk + k0 + 32, tid);
      stage_tile(Bt[nxt], Bblk + k0 + 32, tid);
    }

    const v16bf a0 = frag_a(&At[cur][mwave + lm][0], half);
    const v16bf a1 = frag_a(&At[cur][mwave + 16 + lm][0], half);
#pragma unroll
    for (int t = 0; t < 4; ++t) {
      const v16bf b = frag_b(&Bt[cur][nwave + t * 16 + lm][0], half);
      acc[0][t] = wmma_bf16(a0, b, acc[0][t]);
      acc[1][t] = wmma_bf16(a1, b, acc[1][t]);
    }
  }

#pragma unroll
  for (int t = 0; t < 4; ++t) {
    const int   n  = nbase + t * 16 + lm;
    const float bn = bias[n];
    const int   h  = n >> 6;
    const int   d  = n & 63;
#pragma unroll
    for (int g = 0; g < 2; ++g) {
      const int m0  = mbase + g * 16 + half * 8;  // 8 consecutive rows
      const int bIx = m0 >> 11;                   // whole group in same batch
      const int s0  = m0 & 2047;
      if (mode == 2) {
        // V^T: consecutive rows -> consecutive s -> one packed 16B store
        v8bf pk;
#pragma unroll
        for (int r = 0; r < 8; ++r) pk[r] = (bf16)(acc[g][t][r] + bn);
        *reinterpret_cast<v8bf*>(
            &VTo[(((size_t)bIx * NH + h) * DH + d) * SS + s0]) = pk;
      } else {
        bf16* Out = (mode == 0) ? Qo : Ko;
#pragma unroll
        for (int r = 0; r < 8; ++r) {
          Out[(((size_t)bIx * NH + h) * SS + (s0 + r)) * DH + d] =
              (bf16)(acc[g][t][r] + bn);
        }
      }
    }
  }
}

// ---------------------------------------------------------------------------
// Fused flash attention. grid = (SS/128, B*H). block = 256 (8 waves).
// K/V chunks staged to LDS via async-to-LDS DMA, double-buffered. Scores are
// computed transposed (S^T = K*Q^T) so softmax stays lane-local; O^T = V^T*P^T.
__global__ void __launch_bounds__(256) flash_attn_kernel(
    const bf16* __restrict__ Q, const bf16* __restrict__ K,
    const bf16* __restrict__ VT, bf16* __restrict__ O) {
  __shared__ __align__(16) bf16 Kt[2][32][72];   // 32 k-rows x 64 d (+pad)
  __shared__ __align__(16) bf16 Vt[2][64][LP];   // 64 d-rows x 32 k (+pad)
  __shared__ __align__(16) bf16 Plds[8][16][LP]; // per-wave P tile

  const int tid  = threadIdx.x;
  const int lane = tid & 31;
  const int wv   = tid >> 5;
  const int half = lane >> 4;
  const int lm   = lane & 15;

  const int bh    = blockIdx.y;
  const int bIx   = bh >> 4;
  const int h     = bh & 15;
  const int qbase = blockIdx.x * 128 + wv * 16;

  const bf16* Qb = Q + ((size_t)bh * SS + qbase) * DH;
  const bf16* Kb = K + (size_t)bh * SS * DH;
  const bf16* Vb = VT + (size_t)bh * DH * SS;

  // per-thread staging slots: 256 threads move 256 x 16B per tile
  const int krow = tid >> 3, kc = (tid & 7) * 8;  // K tile: 32 rows x 8 chunks
  const int vrow = tid >> 2, vc = (tid & 3) * 8;  // V tile: 64 rows x 4 chunks

  // stage chunk 0 -> buffer 0
  async_ld16(&Kt[0][krow][kc], Kb + (size_t)krow * DH + kc);
  async_ld16(&Vt[0][vrow][vc], Vb + (size_t)vrow * SS + vc);

  // Q^T as B fragments (lane = q, elements = d), loaded once from global.
  const v16bf qf0 = frag_b(Qb + (size_t)lm * DH, half);       // d 0..31
  const v16bf qf1 = frag_b(Qb + (size_t)lm * DH + 32, half);  // d 32..63

  float m_run = -1e30f;
  float l_run = 0.0f;
  v8f oacc[4];
#pragma unroll
  for (int t = 0; t < 4; ++t) oacc[t] = vzero8();

  for (int kb = 0; kb < SS; kb += 32) {
    const int cur = (kb >> 5) & 1;

    wait_async0();      // chunk `kb` (prefetched last iteration) has landed
    __syncthreads();    // ... for every wave; prior readers of `nxt` done

    if (kb + 32 < SS) {
      const int nxt = cur ^ 1;
      async_ld16(&Kt[nxt][krow][kc], Kb + (size_t)(kb + 32 + krow) * DH + kc);
      async_ld16(&Vt[nxt][vrow][vc], Vb + (size_t)vrow * SS + (kb + 32) + vc);
    }

    // S^T tiles from LDS: (16 k-rows) x (16 q-cols), K-dim = 64 -> 2 WMMA each.
    v8f s0 = vzero8(), s1 = vzero8();
    {
      const bf16* Kr0 = &Kt[cur][lm][0];
      s0 = wmma_bf16(frag_a(Kr0, half), qf0, s0);
      s0 = wmma_bf16(frag_a(Kr0 + 32, half), qf1, s0);
      const bf16* Kr1 = &Kt[cur][16 + lm][0];
      s1 = wmma_bf16(frag_a(Kr1, half), qf0, s1);
      s1 = wmma_bf16(frag_a(Kr1 + 32, half), qf1, s1);
    }

    // Online softmax over this lane's q-row (k = half*8+r and 16+half*8+r).
    const float scale = 0.125f;  // 1/sqrt(64)
    float tmax = -1e30f;
#pragma unroll
    for (int r = 0; r < 8; ++r) {
      s0[r] *= scale;
      s1[r] *= scale;
      tmax = fmaxf(tmax, fmaxf(s0[r], s1[r]));
    }
    tmax = fmaxf(tmax, __shfl_xor(tmax, 16));  // combine the two k-halves
    const float mnew  = fmaxf(m_run, tmax);
    const float alpha = __expf(m_run - mnew);

    float tsum = 0.0f;
#pragma unroll
    for (int r = 0; r < 8; ++r) {
      const float p0 = __expf(s0[r] - mnew);
      const float p1 = __expf(s1[r] - mnew);
      tsum += p0 + p1;
      Plds[wv][lm][half * 8 + r]      = (bf16)p0;  // k_rel = half*8+r
      Plds[wv][lm][16 + half * 8 + r] = (bf16)p1;  // k_rel = 16+half*8+r
    }
    tsum += __shfl_xor(tsum, 16);
    l_run = l_run * alpha + tsum;
    m_run = mnew;

#pragma unroll
    for (int t = 0; t < 4; ++t)
#pragma unroll
      for (int r = 0; r < 8; ++r) oacc[t][r] *= alpha;

    // same-wave LDS store -> load ordering (CDNA5 split counters)
    asm volatile("s_wait_dscnt 0x0" ::: "memory");

    // P^T as B fragment: lane = q, elements = k_rel = half*8+e (e<8), 16+half*8+(e-8) (e>=8).
    const v16bf pf = frag_b(&Plds[wv][lm][0], half);

    // O^T += V^T(d-tile, 32 k) * P^T(32 k, 16 q): 4 d-tiles from LDS.
#pragma unroll
    for (int t = 0; t < 4; ++t) {
      oacc[t] = wmma_bf16(frag_a(&Vt[cur][t * 16 + lm][0], half), pf, oacc[t]);
    }
  }

  // Epilogue: lane holds column q = lm of O^T -> row s of O. Normalize by l.
  const float inv_l = 1.0f / l_run;
  const int   s_idx = qbase + lm;
  bf16* Orow = O + ((size_t)bIx * SS + s_idx) * DM + h * DH;
#pragma unroll
  for (int t = 0; t < 4; ++t) {
#pragma unroll
    for (int r = 0; r < 8; ++r) {
      const int d = t * 16 + half * 8 + r;
      Orow[d] = (bf16)(oacc[t][r] * inv_l);
    }
  }
}

// ---------------------------------------------------------------------------
// Output projection: f32 result. grid = (MT/128, DM/128). block = 256.
// Same LDS-staged 128x128 block tile as the QKV GEMM.
__global__ void __launch_bounds__(256) out_gemm_kernel(
    const bf16* __restrict__ X, const bf16* __restrict__ W,
    const float* __restrict__ bias, float* __restrict__ Out) {
  __shared__ __align__(16) bf16 At[2][128][LP];
  __shared__ __align__(16) bf16 Bt[2][128][LP];

  const int tid  = threadIdx.x;
  const int lane = tid & 31;
  const int wv   = tid >> 5;
  const int half = lane >> 4;
  const int lm   = lane & 15;

  const int mwave = (wv & 3) * 32;
  const int nwave = (wv >> 2) * 64;
  const int mbase = blockIdx.x * 128 + mwave;
  const int nbase = blockIdx.y * 128 + nwave;

  const bf16* Ablk = X + (size_t)(blockIdx.x * 128) * DM;
  const bf16* Bblk = W + (size_t)(blockIdx.y * 128) * DM;

  stage_tile(At[0], Ablk, tid);
  stage_tile(Bt[0], Bblk, tid);

  v8f acc[2][4];
#pragma unroll
  for (int g = 0; g < 2; ++g)
#pragma unroll
    for (int t = 0; t < 4; ++t) acc[g][t] = vzero8();

  for (int k0 = 0; k0 < DM; k0 += 32) {
    const int cur = (k0 >> 5) & 1;

    wait_async0();
    __syncthreads();

    if (k0 + 32 < DM) {
      const int nxt = cur ^ 1;
      stage_tile(At[nxt], Ablk + k0 + 32, tid);
      stage_tile(Bt[nxt], Bblk + k0 + 32, tid);
    }

    const v16bf a0 = frag_a(&At[cur][mwave + lm][0], half);
    const v16bf a1 = frag_a(&At[cur][mwave + 16 + lm][0], half);
#pragma unroll
    for (int t = 0; t < 4; ++t) {
      const v16bf b = frag_b(&Bt[cur][nwave + t * 16 + lm][0], half);
      acc[0][t] = wmma_bf16(a0, b, acc[0][t]);
      acc[1][t] = wmma_bf16(a1, b, acc[1][t]);
    }
  }

#pragma unroll
  for (int t = 0; t < 4; ++t) {
    const int   n  = nbase + t * 16 + lm;
    const float bn = bias[n];
#pragma unroll
    for (int g = 0; g < 2; ++g) {
#pragma unroll
      for (int r = 0; r < 8; ++r) {
        const int m = mbase + g * 16 + half * 8 + r;
        Out[(size_t)m * DM + n] = acc[g][t][r] + bn;
      }
    }
  }
}

// ---------------------------------------------------------------------------
extern "C" void kernel_launch(void* const* d_in, const int* in_sizes, int n_in,
                              void* d_out, int out_size, void* d_ws,
                              size_t ws_size, hipStream_t stream) {
  (void)in_sizes; (void)n_in; (void)out_size; (void)ws_size;

  const float* x  = (const float*)d_in[0];
  const float* Wq = (const float*)d_in[1];
  const float* bq = (const float*)d_in[2];
  const float* Wk = (const float*)d_in[3];
  const float* bk = (const float*)d_in[4];
  const float* Wv = (const float*)d_in[5];
  const float* bv = (const float*)d_in[6];
  const float* Wo = (const float*)d_in[7];
  const float* bo = (const float*)d_in[8];
  float* out = (float*)d_out;

  const size_t SZX = (size_t)MT * DM;  // 8,388,608
  const size_t SZW = (size_t)DM * DM;  // 1,048,576

  bf16* ws  = (bf16*)d_ws;
  bf16* Xb  = ws;
  bf16* Wqb = Xb + SZX;
  bf16* Wkb = Wqb + SZW;
  bf16* Wvb = Wkb + SZW;
  bf16* Wob = Wvb + SZW;
  bf16* Qb  = Wob + SZW;
  bf16* Kb  = Qb + SZX;
  bf16* VTb = Kb + SZX;
  bf16* Ob  = VTb + SZX;  // total ~92 MB of workspace

  // 1) casts
  cast_f32_to_bf16<<<(int)((SZX + 255) / 256), 256, 0, stream>>>(x, Xb, (int)SZX);
  cast_f32_to_bf16<<<(int)((SZW + 255) / 256), 256, 0, stream>>>(Wq, Wqb, (int)SZW);
  cast_f32_to_bf16<<<(int)((SZW + 255) / 256), 256, 0, stream>>>(Wk, Wkb, (int)SZW);
  cast_f32_to_bf16<<<(int)((SZW + 255) / 256), 256, 0, stream>>>(Wv, Wvb, (int)SZW);
  cast_f32_to_bf16<<<(int)((SZW + 255) / 256), 256, 0, stream>>>(Wo, Wob, (int)SZW);

  // 2) QKV projections (z: 0=Q, 1=K, 2=V^T)
  dim3 g1(MT / 128, DM / 128, 3);
  qkv_gemm_kernel<<<g1, 256, 0, stream>>>(Xb, Wqb, Wkb, Wvb, bq, bk, bv,
                                          Qb, Kb, VTb);

  // 3) fused attention
  dim3 g2(SS / 128, BB * NH);
  flash_attn_kernel<<<g2, 256, 0, stream>>>(Qb, Kb, VTb, Ob);

  // 4) output projection
  dim3 g3(MT / 128, DM / 128);
  out_gemm_kernel<<<g3, 256, 0, stream>>>(Ob, Wob, bo, out);
}
